// MolTransformer_65317862637731
// MI455X (gfx1250) — matrix operand order repaired and checked by reading, not compiled, not run on hardware
//
#include <hip/hip_runtime.h>
#include <hip/hip_bf16.h>

// ---- problem constants -----------------------------------------------------
#define N_HEADS 4
#define D_K     32
#define LEAKK   0.2f
#define EPSS    1e-20f
#define BB      8
#define AAA     128
#define F_ATOM  98
#define F_PATH  34
#define HIDDEN  256

typedef __attribute__((ext_vector_type(16))) __bf16 v16bf;
typedef __attribute__((ext_vector_type(8)))  float  v8f;

union BF16x16 { v16bf v; unsigned int w[8]; unsigned short u[16]; };
union F32x8   { v8f   v; float f[8]; };

__device__ __forceinline__ unsigned int f2bf(float x) {
  unsigned int u = __float_as_uint(x);
  u += 0x7FFFu + ((u >> 16) & 1u);          // round-to-nearest-even
  return u >> 16;
}

// ---------------------------------------------------------------------------
// Generic tiled bf16 WMMA GEMM: C[z][M][N] = A[z][M][K] @ B[K][N] (+bias)(relu)
// Block tile: 128(M) x 64(N), K stepped by 32 (zero-padded past K, branchless).
// Requires M % 128 == 0, N % 64 == 0 (true for all call sites here).
// blockDim = 256 (8 waves); wave w owns m-tile w and 4 n-tiles.
// LDS staged as packed bf16 pairs (u32) matching WMMA fragment K-pair order.
// ---------------------------------------------------------------------------
__global__ __launch_bounds__(256)
void molt_gemm_bf16(const float* __restrict__ A, const float* __restrict__ Bw,
                    float* __restrict__ C, const float* __restrict__ bias,
                    int M, int N, int K,
                    long long strideA, long long strideC, int relu)
{
  __shared__ unsigned int lA[128 * 16];     // [m][kpair]  (8 KB)
  __shared__ unsigned int lB[64 * 16];      // [n][kpair]  (4 KB)

  const int tid  = threadIdx.x;
  const int lane = tid & 31;
  const int wave = tid >> 5;
  const int n0 = blockIdx.x * 64;
  const int m0 = blockIdx.y * 128;
  const float* Ab = A + (long long)blockIdx.z * strideA;
  float*       Cb = C + (long long)blockIdx.z * strideC;

  F32x8 acc[4];
  for (int t = 0; t < 4; ++t)
    for (int r = 0; r < 8; ++r) acc[t].f[r] = 0.0f;

  const int half = lane >> 4;               // 0: lanes 0-15, 1: lanes 16-31
  const int l15  = lane & 15;

  for (int k0 = 0; k0 < K; k0 += 32) {
    // stage A tile (128x32 bf16 = 2048 u32), branchless zero-pad in K
    #pragma unroll
    for (int r = 0; r < 8; ++r) {
      int p  = tid + r * 256;               // 0..2047
      int mm = p >> 4, kp = p & 15;
      int kk = k0 + 2 * kp;
      const float* base = Ab + (long long)(m0 + mm) * K;
      float x0 = base[(kk     < K) ? kk     : 0];
      float x1 = base[(kk + 1 < K) ? kk + 1 : 0];
      if (kk     >= K) x0 = 0.0f;
      if (kk + 1 >= K) x1 = 0.0f;
      lA[p] = f2bf(x0) | (f2bf(x1) << 16);
    }
    // stage B tile (32x64 bf16 = 1024 u32), transposed to [n][kpair]
    #pragma unroll
    for (int r = 0; r < 4; ++r) {
      int p  = tid + r * 256;               // 0..1023
      int nn = p >> 4, kp = p & 15;
      int kk = k0 + 2 * kp;
      float x0 = Bw[(long long)((kk     < K) ? kk     : 0) * N + n0 + nn];
      float x1 = Bw[(long long)((kk + 1 < K) ? kk + 1 : 0) * N + n0 + nn];
      if (kk     >= K) x0 = 0.0f;
      if (kk + 1 >= K) x1 = 0.0f;
      lB[p] = f2bf(x0) | (f2bf(x1) << 16);
    }
    __syncthreads();

    // A fragment (16-bit A 16x32 layout, ISA 7.12.2), read as K-pair dwords
    BF16x16 af;
    const int mrow = (wave << 4) + l15;
    #pragma unroll
    for (int p = 0; p < 8; ++p) {
      int kp = (p < 4) ? (p + 4 * half) : (p + 4 + 4 * half);
      af.w[p] = lA[mrow * 16 + kp];
    }
    #pragma unroll
    for (int t = 0; t < 4; ++t) {
      BF16x16 bf;
      const int ncol = t * 16 + l15;        // B fragment: same K order as A fragment
      #pragma unroll
      for (int p = 0; p < 8; ++p)
        bf.w[p] = lB[ncol * 16 + ((p < 4) ? (p + 4 * half) : (p + 4 + 4 * half))];
      acc[t].v = __builtin_amdgcn_wmma_f32_16x16x32_bf16(
          false, af.v, false, bf.v, (short)0, acc[t].v, false, false);
    }
    __syncthreads();
  }

  // epilogue: D layout (ISA 7.12.2): VGPR r -> M = r + 8*half, N = lane%16
  #pragma unroll
  for (int t = 0; t < 4; ++t) {
    const int ncol = n0 + t * 16 + l15;
    const float bv = bias ? bias[ncol] : 0.0f;
    #pragma unroll
    for (int r = 0; r < 8; ++r) {
      const int mrow = m0 + (wave << 4) + r + 8 * half;
      float x = acc[t].f[r] + bv;
      if (relu) x = fmaxf(x, 0.0f);
      Cb[(long long)mrow * N + ncol] = x;
    }
  }
}

// ---------------------------------------------------------------------------
// Weight prep: Wcat[32][128] = [W1|W2|M1|M2], W34[34][64] = [W3|M3]
// ---------------------------------------------------------------------------
__global__ void molt_prep_w(const float* __restrict__ W_attn_h,
                            const float* __restrict__ W_msg_h,
                            float* __restrict__ Wcat, float* __restrict__ W34)
{
  int idx = blockIdx.x * 256 + threadIdx.x;
  if (idx < 32 * 128) {
    int k = idx >> 7, c = idx & 127;
    float v;
    if (c < 32)       v = W_attn_h[k * 32 + c];
    else if (c < 64)  v = W_attn_h[(32 + k) * 32 + (c - 32)];
    else if (c < 96)  v = W_msg_h [k * 32 + (c - 64)];
    else              v = W_msg_h [(32 + k) * 32 + (c - 96)];
    Wcat[idx] = v;
  } else if (idx < 32 * 128 + 34 * 64) {
    int j = idx - 32 * 128;
    int k = j >> 6, c = j & 63;
    W34[j] = (c < 32) ? W_attn_h[(64 + k) * 32 + c]
                      : W_msg_h [(64 + k) * 32 + (c - 32)];
  }
}

// [B,A,H*32] -> [H,B,A,32]
__global__ void molt_reorder(const float* __restrict__ AIHflat,
                             float* __restrict__ AIHh)
{
  int idx = blockIdx.x * 256 + threadIdx.x;          // 131072 threads
  int d = idx & 31; int t = idx >> 5;
  int i = t & 127;  t >>= 7;
  int b = t & 7;    int h = t >> 3;
  AIHh[idx] = AIHflat[(b * AAA + i) * (N_HEADS * D_K) + h * D_K + d];
}

// CAT[1024][226] = [atom_input | atom_h(H,B,A,32 -> B,A,H*32)]
__global__ void molt_concat(const float* __restrict__ atom,
                            const float* __restrict__ atomh,
                            float* __restrict__ CAT)
{
  int idx = blockIdx.x * 256 + threadIdx.x;
  if (idx >= 1024 * 226) return;
  int row = idx / 226, c = idx % 226;
  float v;
  if (c < F_ATOM) {
    v = atom[row * F_ATOM + c];
  } else {
    int cc = c - F_ATOM;
    int hh = cc >> 5, dd = cc & 31;
    int bb = row >> 7, ii = row & 127;
    v = atomh[(((size_t)hh * BB + bb) * AAA + ii) * D_K + dd];
  }
  CAT[idx] = v;
}

// ---------------------------------------------------------------------------
// Fused attention layer with online (flash-style) masked softmax.
// grid = (A/8, B, H), block = 256 (8 waves); wave w handles row i, lane = d_k.
// PQ[hb][j][0:32]=P1, [32:64]=P2, [64:96]=Q1, [96:128]=Q2
// P3PM3[b][i][j][0:32]=path@W3, [32:64]=path@M3
// ---------------------------------------------------------------------------
__global__ __launch_bounds__(256)
void molt_attn_layer(const float* __restrict__ PQ,
                     const float* __restrict__ P3PM3,
                     const float* __restrict__ mask,
                     const float* __restrict__ AIHh,
                     const float* __restrict__ b_attn_h,
                     const float* __restrict__ W_attn_o,
                     const float* __restrict__ b_attn_o,
                     const float* __restrict__ b_msg_h,
                     float* __restrict__ out)
{
  __shared__ float lP2[AAA * D_K];    // 16 KB
  __shared__ float lQ2[AAA * D_K];    // 16 KB
  const int tid = threadIdx.x;
  const int lane = tid & 31, wave = tid >> 5;
  const int b = blockIdx.y, h = blockIdx.z;
  const int hb = h * BB + b;
  const float* pq = PQ + (size_t)hb * AAA * 128;

  for (int r = tid; r < AAA * D_K; r += 256) {
    int j = r >> 5, d2 = r & 31;
    const float* row = pq + (size_t)j * 128;
    lP2[r] = row[32 + d2];
    lQ2[r] = row[96 + d2];
  }
  __syncthreads();

  const int i = blockIdx.x * 8 + wave;
  const int d = lane;
  const float* prow = pq + (size_t)i * 128;
  const float P1d = prow[d];
  const float Q1d = prow[64 + d];
  const float ba  = b_attn_h[d];
  const float wo  = W_attn_o[d];
  const float bo  = b_attn_o[0];
  const float bm  = b_msg_h[d];
  const float res = AIHh[((size_t)hb * AAA + i) * D_K + d];
  const float* mrow  = mask + ((size_t)b * AAA + i) * AAA;
  const float* p3row = P3PM3 + (size_t)(b * AAA + i) * AAA * 64;

  float m = -3.0e38f, l = 0.f, accw = 0.f, a2 = 0.f, a3 = 0.f;
  for (int j = 0; j < AAA; ++j) {
    const float* pj = p3row + (size_t)j * 64;
    __builtin_prefetch(pj + 8 * 64, 0, 1);          // global_prefetch_b8
    float p3  = pj[d];
    float pm3 = pj[32 + d];
    float s = P1d + lP2[(j << 5) + d] + p3 + ba;
    s = (s > 0.f) ? s : LEAKK * s;                  // leaky_relu
    float red = s * wo;                             // dot(s, W_attn_o) across lanes
    #pragma unroll
    for (int off = 16; off > 0; off >>= 1) red += __shfl_xor(red, off, 32);
    float mk = mrow[j];
    float sc = (red + bo) * mk;
    // online masked softmax update
    float mn = fmaxf(m, sc);
    float scale = __expf(m - mn);
    float e = __expf(sc - mn) * mk;                 // e = exp(score-max)*mask
    float w = e * mk;                               // probs numerator incl. 2nd mask
    l    = l    * scale + e;
    accw = accw * scale + w;                        // -> rowsum(probs)
    a2   = a2   * scale + w * lQ2[(j << 5) + d];    // probs @ (h @ M2)
    a3   = a3   * scale + w * pm3;                  // sum_j probs * (path @ M3)
    m = mn;
  }
  float inv = 1.0f / (l + EPSS);
  float val = accw * inv * Q1d + a2 * inv + a3 * inv + bm + res;
  out[((size_t)hb * AAA + i) * D_K + d] = fmaxf(val, 0.0f);
}

// ---------------------------------------------------------------------------
extern "C" void kernel_launch(void* const* d_in, const int* in_sizes, int n_in,
                              void* d_out, int out_size, void* d_ws, size_t ws_size,
                              hipStream_t stream) {
  (void)in_sizes; (void)n_in; (void)out_size; (void)ws_size;
  const float* atom     = (const float*)d_in[0];   // [8,128,98]
  const float* path     = (const float*)d_in[1];   // [8,128,128,34]
  const float* pmask    = (const float*)d_in[2];   // [8,128,128]
  const float* W_atom_i = (const float*)d_in[3];   // [98,128]
  const float* W_attn_h = (const float*)d_in[4];   // [98,32]
  const float* b_attn_h = (const float*)d_in[5];   // [32]
  const float* W_attn_o = (const float*)d_in[6];   // [32,1]
  const float* b_attn_o = (const float*)d_in[7];   // [1]
  const float* W_msg_h  = (const float*)d_in[8];   // [98,32]
  const float* b_msg_h  = (const float*)d_in[9];   // [32]
  const float* W_atom_o = (const float*)d_in[10];  // [226,256]
  const float* b_atom_o = (const float*)d_in[11];  // [256]
  float* out = (float*)d_out;                      // [1024,256]

  // workspace layout (floats)
  float* ws = (float*)d_ws;
  size_t off = 0;
  float* AIHflat = ws + off; off += (size_t)1024 * 128;            // [B,A,128]
  float* AIHh    = ws + off; off += (size_t)N_HEADS * BB * AAA * D_K;
  float* atomh0  = ws + off; off += (size_t)N_HEADS * BB * AAA * D_K;
  float* atomh1  = ws + off; off += (size_t)N_HEADS * BB * AAA * D_K;
  float* PQ      = ws + off; off += (size_t)32 * AAA * 128;        // [32][128][128]
  float* P3PM3   = ws + off; off += (size_t)BB * AAA * AAA * 64;   // 33.6 MB
  float* Wcat    = ws + off; off += (size_t)32 * 128;
  float* W34     = ws + off; off += (size_t)34 * 64;
  float* CAT     = ws + off; off += (size_t)1024 * 226;

  // 1) weight prep
  molt_prep_w<<<(32 * 128 + 34 * 64 + 255) / 256, 256, 0, stream>>>(
      W_attn_h, W_msg_h, Wcat, W34);

  // 2) AIHflat = atom @ W_atom_i   (M=1024, N=128, K=98)
  molt_gemm_bf16<<<dim3(2, 8, 1), 256, 0, stream>>>(
      atom, W_atom_i, AIHflat, nullptr, 1024, 128, F_ATOM, 0, 0, 0);

  // 3) P3PM3 = path @ [W3|M3]      (M=8*128*128=131072, N=64, K=34)  — path read ONCE
  molt_gemm_bf16<<<dim3(1, 1024, 1), 256, 0, stream>>>(
      path, W34, P3PM3, nullptr, BB * AAA * AAA, 64, F_PATH, 0, 0, 0);

  // 4) reorder projection to [H,B,A,32] (initial atom_h and residual)
  molt_reorder<<<(N_HEADS * BB * AAA * D_K) / 256, 256, 0, stream>>>(AIHflat, AIHh);

  // 5) two shared-weight layers
  const float* cur = AIHh;
  float* bufs[2] = {atomh0, atomh1};
  for (int layer = 0; layer < 2; ++layer) {
    // batched per-(h,b): [P1|P2|Q1|Q2] = atom_h @ Wcat (M=128, N=128, K=32, 32 batches)
    molt_gemm_bf16<<<dim3(2, 1, 32), 256, 0, stream>>>(
        cur, Wcat, PQ, nullptr, AAA, 128, D_K,
        (long long)AAA * D_K, (long long)AAA * 128, 0);
    molt_attn_layer<<<dim3(AAA / 8, BB, N_HEADS), 256, 0, stream>>>(
        PQ, P3PM3, pmask, AIHh, b_attn_h, W_attn_o, b_attn_o, b_msg_h,
        bufs[layer]);
    cur = bufs[layer];
  }

  // 6) concat + output GEMM: out = relu(CAT @ W_atom_o + b) (M=1024, N=256, K=226)
  molt_concat<<<(1024 * 226 + 255) / 256, 256, 0, stream>>>(atom, cur, CAT);
  molt_gemm_bf16<<<dim3(4, 8, 1), 256, 0, stream>>>(
      CAT, W_atom_o, out, b_atom_o, 1024, HIDDEN, F_ATOM + N_HEADS * D_K, 0, 0, 1);
}
